// OctreeConvRelu_63651415327310
// MI455X (gfx1250) — matrix operand order repaired and checked by reading, not compiled, not run on hardware
//
#include <hip/hip_runtime.h>
#include <hip/hip_bf16.h>

typedef __attribute__((ext_vector_type(16))) _Float16 v16h;
typedef __attribute__((ext_vector_type(8)))  _Float16 v8h;
typedef __attribute__((ext_vector_type(8)))  float    v8f;

#define C_IN   64
#define C_OUT  128
#define KSZ    27
#define KDIM   (C_IN * KSZ)          // 1728
#define KCHUNK 32                    // WMMA K per step
#define NKC    (KDIM / KCHUNK)       // 54
#define NT     16                    // h-columns per workgroup
#define NWAVE  4                     // 4 waves, each owns 2 M-tiles (2*16 rows)
#define BLOCK  (NWAVE * 32)          // 128 threads
#define MT     (C_OUT / 16)          // 8 m-tiles total
#define LDS_PAD 8
#define LDS_ROW (KDIM + LDS_PAD)     // 1736 halfs/row: 16B-aligned rows, 36-dword bank stride

// --- Kernel 1: transpose + f32->f16 convert:  xT[h*64+c] = x[c*H+h] ---
__global__ void k_transpose_x(const float* __restrict__ x,
                              _Float16* __restrict__ xT, int H) {
    int t = blockIdx.x * blockDim.x + threadIdx.x;
    if (t >= C_IN * H) return;
    int h = t >> 6;
    int c = t & 63;
    xT[t] = (_Float16)x[(size_t)c * H + h];
}

// --- Kernel 2: pack weight into per-lane WMMA A-fragment order (f16) ---
// Wp[((mt*NKC + kc)*32 + lane)*16 + e] = W[M=mt*16+lane%16, kk=kc*32+off]
// A 16-bit 16x32 layout (ISA 7.12.2): lane<16: K {0..7,16..23}; lane>=16: K {8..15,24..31}
__global__ void k_pack_w(const float* __restrict__ W,
                         _Float16* __restrict__ Wp, int total) {
    int t = blockIdx.x * blockDim.x + threadIdx.x;
    if (t >= total) return;
    int e    = t & 15;
    int lane = (t >> 4) & 31;
    int kc   = (t >> 9) % NKC;
    int mt   = (t >> 9) / NKC;
    int M    = mt * 16 + (lane & 15);
    int off;
    if (lane < 16) off = (e < 8) ? e       : (8 + e);
    else           off = (e < 8) ? (8 + e) : (16 + e);
    int kk = kc * KCHUNK + off;
    int c  = kk & 63;                 // kk = k*64 + c (c fastest)
    int k  = kk >> 6;
    Wp[t] = (_Float16)W[((size_t)M * C_IN + c) * KSZ + k];
}

// --- Kernel 3: async gather into LDS + WMMA GEMM + ReLU ---
__global__ __launch_bounds__(BLOCK) void k_octconv_wmma(
        const _Float16* __restrict__ xT,
        const int*      __restrict__ neigh,
        const _Float16* __restrict__ Wp,
        float*          __restrict__ out, int H) {
    __shared__ _Float16 tile[NT][LDS_ROW];

    const int h0  = blockIdx.x * NT;
    const int tid = threadIdx.x;

    // Cooperative gather: NT*27 slices of 64 halfs, in 16B units. units = NT*27*8 = 3456.
    // CDNA5 path: global_load_async_to_lds_b128 (per-lane 16B global->LDS, ASYNCcnt).
    for (int u = tid; u < NT * KSZ * 8; u += BLOCK) {
        int part = u & 7;
        int g    = u >> 3;
        int h    = g / KSZ;
        int k    = g % KSZ;
        int hh   = h0 + h;
        int j    = (hh < H) ? neigh[(size_t)hh * KSZ + k] : -1;
        _Float16* dst = &tile[h][k * C_IN + part * 8];
        if (j >= 0) {
            const _Float16* src = xT + (size_t)j * C_IN + part * 8;
            unsigned lds_off = (unsigned)(uintptr_t)dst;   // flat->LDS uses addr[31:0]
            asm volatile("global_load_async_to_lds_b128 %0, %1, off"
                         :: "v"(lds_off), "v"(src) : "memory");
        } else {
            *(v8h*)dst = (v8h){};                          // reference mask: neigh<0 -> 0
        }
    }
    asm volatile("s_wait_asynccnt 0x0" ::: "memory");
    __syncthreads();

    const int wave = tid >> 5;          // owns M-tiles 2*wave, 2*wave+1
    const int lane = tid & 31;
    const int hcol = lane & 15;         // N column
    const int bsel = (lane >> 4) * 8;   // B layout: lanes>=16 take K 8..15,24..31

    v8f acc0 = {};
    v8f acc1 = {};
    const _Float16* arow0 = Wp + ((size_t)(2 * wave) * NKC * 32 + lane) * 16;
    const _Float16* arow1 = arow0 + (size_t)NKC * 32 * 16;   // next M-tile
    const _Float16* brow  = &tile[hcol][bsel];

    #pragma unroll 2
    for (int kc = 0; kc < NKC; ++kc) {
        v8h blo = *(const v8h*)(brow + kc * KCHUNK);
        v8h bhi = *(const v8h*)(brow + kc * KCHUNK + 16);
        v16h b = __builtin_shufflevector(blo, bhi,
                 0,1,2,3,4,5,6,7,8,9,10,11,12,13,14,15);
        v16h a0 = *(const v16h*)(arow0 + (size_t)kc * 512);  // 32B contiguous per lane
        v16h a1 = *(const v16h*)(arow1 + (size_t)kc * 512);
        acc0 = __builtin_amdgcn_wmma_f32_16x16x32_f16(
                   false, a0, false, b, (short)0, acc0, false, false);
        acc1 = __builtin_amdgcn_wmma_f32_16x16x32_f16(
                   false, a1, false, b, (short)0, acc1, false, false);
    }

    // D layout: lane<16 -> (M=r, N=lane); lane>=16 -> (M=8+r, N=lane-16)
    const int n = h0 + hcol;
    if (n < H) {
        const int mbase0 = (2 * wave) * 16 + (lane >> 4) * 8;
        #pragma unroll
        for (int r = 0; r < 8; ++r) {
            float v0 = acc0[r];
            out[(size_t)(mbase0 + r) * H + n] = v0 > 0.0f ? v0 : 0.0f;
        }
        #pragma unroll
        for (int r = 0; r < 8; ++r) {
            float v1 = acc1[r];
            out[(size_t)(mbase0 + 16 + r) * H + n] = v1 > 0.0f ? v1 : 0.0f;
        }
    }
}

extern "C" void kernel_launch(void* const* d_in, const int* in_sizes, int n_in,
                              void* d_out, int out_size, void* d_ws, size_t ws_size,
                              hipStream_t stream) {
    const float* x     = (const float*)d_in[0];   // (1,64,H,1) f32
    const int*   neigh = (const int*)  d_in[1];   // (H,27) int32
    const float* W     = (const float*)d_in[2];   // (128,64,27) f32
    float*       out   = (float*)d_out;           // (1,128,H,1) f32

    const int H = in_sizes[0] / C_IN;

    _Float16* xT = (_Float16*)d_ws;
    _Float16* Wp = (_Float16*)((char*)d_ws + (size_t)C_IN * H * sizeof(_Float16));

    const int totalX = C_IN * H;
    k_transpose_x<<<(totalX + 255) / 256, 256, 0, stream>>>(x, xT, H);

    const int totalW = MT * NKC * 32 * 16;   // 221184
    k_pack_w<<<(totalW + 255) / 256, 256, 0, stream>>>(W, Wp, totalW);

    k_octconv_wmma<<<(H + NT - 1) / NT, BLOCK, 0, stream>>>(xT, neigh, Wp, out, H);
}
